// AttentionFlow_78228534329586
// MI455X (gfx1250) — matrix operand order repaired and checked by reading, not compiled, not run on hardware
//
#include <hip/hip_runtime.h>
#include <hip/hip_bf16.h>
#include <stdint.h>

typedef __bf16 bf16_t;
typedef __attribute__((ext_vector_type(16))) __bf16 v16bf;
typedef __attribute__((ext_vector_type(8)))  __bf16 v8bf;
typedef __attribute__((ext_vector_type(8)))  float  v8f;

#define N_NODES 100000
#define DIM     256
#define BQ      64
#define EPGC    1024
#define NE      65536
#define KTOP    256

// att_score tile: 32 edges per block (two 16-row WMMA A-tiles -> B frags reused x2)
#define MT      32

// dynamic LDS layout for att_score (bytes)
#define SMEM_SA    0                       // 32 x 1024 bf16 = 65536
#define SMEM_SRT   (MT * 1024 * 2)         // 32 x 512  bf16 = 32768
#define SMEM_SLOG  (SMEM_SRT + MT * 512 * 2)  // 32 floats
#define SMEM_SHJ   (SMEM_SLOG + 128)       // 32 x 256 f32 staging = 32768
#define SMEM_TOTAL (SMEM_SHJ + MT * 256 * 4)  // 131200 B -> 2 workgroups / WGP

// ---------- helpers ----------

__device__ __forceinline__ bf16_t to_bf(float f) {
  unsigned u = __builtin_bit_cast(unsigned, f);
  u += 0x7FFFu + ((u >> 16) & 1u);           // round-to-nearest-even
  unsigned short h = (unsigned short)(u >> 16);
  return __builtin_bit_cast(bf16_t, h);
}

// monotonic float<->uint mapping so atomicMax(uint) == float max
__device__ __forceinline__ unsigned f2mono(float f) {
  unsigned u = __builtin_bit_cast(unsigned, f);
  return ((int)u < 0) ? ~u : (u | 0x80000000u);
}
__device__ __forceinline__ float mono2f(unsigned u) {
  unsigned b = (u & 0x80000000u) ? (u & 0x7FFFFFFFu) : ~u;
  return __builtin_bit_cast(float, b);
}

// A-fragment (16x32 bf16, MxK) from row-major LDS.
__device__ __forceinline__ v16bf load_a_frag(const bf16_t* lds, int lda, int k0, int lane) {
  int m = lane & 15, half = lane >> 4;
  const bf16_t* p = lds + m * lda + k0 + 8 * half;
  union { v16bf v; v8bf h[2]; } u;
  u.h[0] = *(const v8bf*)(p);
  u.h[1] = *(const v8bf*)(p + 16);
  return u.v;
}

// B-fragment (32x16 bf16, KxN) from transposed weights WT[n][k] (row-major, stride kdim).
__device__ __forceinline__ v16bf load_b_frag(const bf16_t* wt, int kdim, int n0, int k0, int lane) {
  const bf16_t* p = wt + (size_t)(n0 + (lane & 15)) * kdim + k0 + 8 * (lane >> 4);
  union { v16bf v; v8bf h[2]; } u;
  u.h[0] = *(const v8bf*)(p);
  u.h[1] = *(const v8bf*)(p + 16);
  return u.v;
}

__device__ __forceinline__ v8f wmma_bf16(v16bf a, v16bf b, v8f c) {
  return __builtin_amdgcn_wmma_f32_16x16x32_bf16(false, a, false, b, (short)0, c, false, false);
}

// gfx1250 async global->LDS copy (16B per lane), tracked by ASYNCcnt
__device__ __forceinline__ void async_copy16(unsigned ldsOff, const void* gptr) {
  unsigned long long ga = (unsigned long long)(uintptr_t)gptr;
  asm volatile("global_load_async_to_lds_b128 %0, %1, off"
               :: "v"(ldsOff), "v"(ga) : "memory");
}
__device__ __forceinline__ void wait_async0() {
  asm volatile("s_wait_asynccnt 0x0" ::: "memory");
}

// ---------- weight convert + transpose (once) ----------

__global__ void cvt_transpose_kernel(const float* __restrict__ W, bf16_t* __restrict__ WT,
                                     int K, int Nn) {
  int idx = blockIdx.x * blockDim.x + threadIdx.x;
  if (idx >= K * Nn) return;
  int n = idx / K, k = idx - n * K;
  WT[idx] = to_bf(W[(size_t)k * Nn + n]);   // WT[n][k] = W[k][n]
}

// ---------- fused gather + 3 chained GEMMs + dot: logits per edge ----------

__global__ __launch_bounds__(256)
void att_score_kernel(const int* __restrict__ edges,
                      const float* __restrict__ rel,
                      const float* __restrict__ emb,
                      const float* __restrict__ qs,
                      const float* __restrict__ qr,
                      const bf16_t* __restrict__ WlT, const float* __restrict__ bl,
                      const bf16_t* __restrict__ WrT, const float* __restrict__ br,
                      const bf16_t* __restrict__ WcT, const float* __restrict__ bc,
                      float* __restrict__ logits) {
  extern __shared__ __align__(16) char smem[];
  bf16_t* sA   = (bf16_t*)(smem + SMEM_SA);   // gathered MTx1024 activation rows
  bf16_t* sRt  = (bf16_t*)(smem + SMEM_SRT);  // intermediate Rt (MTx512)
  float*  sLog = (float*)(smem + SMEM_SLOG);
  float*  sHj  = (float*)(smem + SMEM_SHJ);   // async-staged hj rows (MTx256 f32)

  const int tid  = threadIdx.x;
  const int lane = tid & 31;
  const int wave = tid >> 5;
  const int eBase = blockIdx.x * MT;

  // 8 threads per row; each fills 128 contiguous cols (within one 256-wide segment)
  // hjStage != nullptr: take segment 0 (node embedding) from LDS staging instead of global
  auto fill = [&](bool useJ, const float* hjStage) {
    int row = tid >> 3;
    int c0  = (tid & 7) * 128;
    int e   = eBase + row;
    int g   = edges[e * 8 + 0];
    int seg = c0 >> 8, off = c0 & 255;
    const float* src;
    if (seg == 0) {
      if (hjStage) {
        src = hjStage + row * DIM + off;
      } else {
        int nd = edges[e * 8 + (useJ ? 7 : 6)];
        src = emb + (size_t)nd * DIM + off;
      }
    } else if (seg == 1) src = rel + (size_t)e * DIM + off;
    else if (seg == 2)   src = qs + (size_t)g * DIM + off;
    else                 src = qr + (size_t)g * DIM + off;
    bf16_t* dst = sA + row * 1024 + c0;
    #pragma unroll
    for (int i = 0; i < 128; i += 4) {
      float4 v = *(const float4*)(src + i);
      dst[i+0] = to_bf(v.x); dst[i+1] = to_bf(v.y);
      dst[i+2] = to_bf(v.z); dst[i+3] = to_bf(v.w);
    }
  };

  v8f Lacc[2][4];

  // ---- phase 1: L = lrelu(left @ Wl + bl), kept in VGPRs ----
  fill(false, nullptr);
  __syncthreads();

  // kick off async prefetch of hj rows into LDS staging while phase 1 computes
  {
    const unsigned hjBase = (unsigned)(uintptr_t)(void*)sHj;
    int f0 = tid * 32;                       // this thread's 32 floats of the MTx256 tile
    #pragma unroll
    for (int i = 0; i < 8; ++i) {
      int f = f0 + i * 4;
      int row = f >> 8, col = f & 255;
      int nd = edges[(eBase + row) * 8 + 7];
      async_copy16(hjBase + (unsigned)f * 4u, emb + (size_t)nd * DIM + col);
    }
  }

  #pragma unroll
  for (int nt = 0; nt < 4; ++nt) {
    const int n0 = (wave * 4 + nt) * 16;
    v8f acc0 = {}, acc1 = {};
    for (int k0 = 0; k0 < 1024; k0 += 32) {
      v16bf b  = load_b_frag(WlT, 1024, n0, k0, lane);
      v16bf a0 = load_a_frag(sA, 1024, k0, lane);
      v16bf a1 = load_a_frag(sA + 16 * 1024, 1024, k0, lane);
      acc0 = wmma_bf16(a0, b, acc0);
      acc1 = wmma_bf16(a1, b, acc1);
    }
    float bv = bl[n0 + (lane & 15)];
    #pragma unroll
    for (int r = 0; r < 8; ++r) {
      float x0 = acc0[r] + bv; acc0[r] = x0 > 0.f ? x0 : 0.01f * x0;
      float x1 = acc1[r] + bv; acc1[r] = x1 > 0.f ? x1 : 0.01f * x1;
    }
    Lacc[0][nt] = acc0;
    Lacc[1][nt] = acc1;
  }
  wait_async0();          // hj staging complete (this wave's issues)
  __syncthreads();        // all waves done reading sA + all async data visible

  // ---- phase 2: Rt = lrelu(right @ Wr + br) -> LDS bf16 ----
  fill(true, sHj);
  if (tid < MT) sLog[tid] = 0.f;
  __syncthreads();
  #pragma unroll
  for (int nt = 0; nt < 4; ++nt) {
    const int n0 = (wave * 4 + nt) * 16;
    v8f acc0 = {}, acc1 = {};
    for (int k0 = 0; k0 < 1024; k0 += 32) {
      v16bf b  = load_b_frag(WrT, 1024, n0, k0, lane);
      v16bf a0 = load_a_frag(sA, 1024, k0, lane);
      v16bf a1 = load_a_frag(sA + 16 * 1024, 1024, k0, lane);
      acc0 = wmma_bf16(a0, b, acc0);
      acc1 = wmma_bf16(a1, b, acc1);
    }
    float bv = br[n0 + (lane & 15)];
    int n  = n0 + (lane & 15);
    int mb = (lane >> 4) * 8;
    #pragma unroll
    for (int r = 0; r < 8; ++r) {
      float x0 = acc0[r] + bv; x0 = x0 > 0.f ? x0 : 0.01f * x0;
      float x1 = acc1[r] + bv; x1 = x1 > 0.f ? x1 : 0.01f * x1;
      sRt[(mb + r) * 512 + n]      = to_bf(x0);
      sRt[(16 + mb + r) * 512 + n] = to_bf(x1);
    }
  }
  __syncthreads();

  // ---- phase 3: R = Rt @ Wc + bc ; partial dot with L; ds_add reduce ----
  float part0[8] = {0,0,0,0,0,0,0,0};
  float part1[8] = {0,0,0,0,0,0,0,0};
  #pragma unroll
  for (int nt = 0; nt < 4; ++nt) {
    const int n0 = (wave * 4 + nt) * 16;
    v8f acc0 = {}, acc1 = {};
    for (int k0 = 0; k0 < 512; k0 += 32) {
      v16bf b  = load_b_frag(WcT, 512, n0, k0, lane);
      v16bf a0 = load_a_frag(sRt, 512, k0, lane);
      v16bf a1 = load_a_frag(sRt + 16 * 512, 512, k0, lane);
      acc0 = wmma_bf16(a0, b, acc0);
      acc1 = wmma_bf16(a1, b, acc1);
    }
    float bv = bc[n0 + (lane & 15)];
    #pragma unroll
    for (int r = 0; r < 8; ++r) {
      part0[r] += Lacc[0][nt][r] * (acc0[r] + bv);
      part1[r] += Lacc[1][nt][r] * (acc1[r] + bv);
    }
  }
  const int mb = (lane >> 4) * 8;
  #pragma unroll
  for (int r = 0; r < 8; ++r) {
    atomicAdd(&sLog[mb + r], part0[r]);
    atomicAdd(&sLog[16 + mb + r], part1[r]);
  }
  __syncthreads();
  if (tid < MT) logits[eBase + tid] = sLog[tid];
}

// ---------- segment softmax passes ----------

__global__ void seg_max_kernel(const float* __restrict__ lg, const int* __restrict__ edges,
                               unsigned* __restrict__ m, int E) {
  int e = blockIdx.x * blockDim.x + threadIdx.x;
  if (e >= E) return;
  atomicMax(&m[edges[e * 8 + 6]], f2mono(lg[e]));
}

__global__ void seg_expsum_kernel(const float* __restrict__ lg, const int* __restrict__ edges,
                                  const unsigned* __restrict__ m, float* __restrict__ eb,
                                  float* __restrict__ ss, int E) {
  int e = blockIdx.x * blockDim.x + threadIdx.x;
  if (e >= E) return;
  int seg = edges[e * 8 + 6];
  float ev = __expf(lg[e] - mono2f(m[seg]));
  eb[e] = ev;
  atomicAdd(&ss[seg], ev);
}

__global__ void seg_norm_kernel(const float* __restrict__ eb, const int* __restrict__ edges,
                                const float* __restrict__ ss, float* __restrict__ sm, int E) {
  int e = blockIdx.x * blockDim.x + threadIdx.x;
  if (e >= E) return;
  sm[e] = eb[e] / ss[edges[e * 8 + 6]];
}

__global__ void ta_kernel(const float* __restrict__ sm, const int* __restrict__ edges,
                          const float* __restrict__ score, float* __restrict__ ta, int E) {
  int e = blockIdx.x * blockDim.x + threadIdx.x;
  if (e >= E) return;
  ta[e] = sm[e] * score[edges[e * 8 + 6]];
}

__global__ void count_kernel(const int* __restrict__ edges, int* __restrict__ cnt, int E) {
  int e = blockIdx.x * blockDim.x + threadIdx.x;
  if (e >= E) return;
  atomicAdd(&cnt[edges[e * 8 + 7]], 1);
}

// ---------- per-group top-256 of 1024 via LDS bitonic sort ----------

__global__ __launch_bounds__(256)
void topk_kernel(const float* __restrict__ ta, const float* __restrict__ sm,
                 const int* __restrict__ edges,
                 float* __restrict__ outScore, float* __restrict__ outEdges,
                 float* __restrict__ outOrig, int* __restrict__ oiList,
                 int* __restrict__ cnt) {
  __shared__ float v[1024];
  __shared__ int   ix[1024];
  const int g = blockIdx.x, tid = threadIdx.x;
  for (int j = tid; j < 1024; j += 256) { v[j] = ta[g * EPGC + j]; ix[j] = j; }
  __syncthreads();
  for (int ksz = 2; ksz <= 1024; ksz <<= 1) {
    for (int jsz = ksz >> 1; jsz > 0; jsz >>= 1) {
      for (int i = tid; i < 1024; i += 256) {
        int l = i ^ jsz;
        if (l > i) {
          bool desc = ((i & ksz) == 0);           // overall descending
          bool sw = desc ? (v[i] < v[l]) : (v[i] > v[l]);
          if (sw) {
            float tv = v[i]; v[i] = v[l]; v[l] = tv;
            int   ti = ix[i]; ix[i] = ix[l]; ix[l] = ti;
          }
        }
      }
      __syncthreads();
    }
  }
  int oi  = g * EPGC + ix[tid];                    // tid in [0,256): top-256
  float pa  = v[tid];
  float smv = sm[oi];
  oiList[g * KTOP + tid] = oi;
  outOrig[g * KTOP + tid] = (float)oi;
  #pragma unroll
  for (int c = 0; c < 8; ++c)
    outEdges[((size_t)(g * KTOP + tid)) * 8 + c] = (float)edges[(size_t)oi * 8 + c];
  int nj = edges[oi * 8 + 7];
  atomicAdd(&outScore[nj], smv * pa);
  atomicAdd(&cnt[nj], 1);
}

// ---------- node update: init + row-scatter ----------

__global__ void emb_init_kernel(float* __restrict__ dst, const float* __restrict__ src,
                                const int* __restrict__ cnt, size_t total) {
  size_t idx = (size_t)blockIdx.x * blockDim.x + threadIdx.x;
  if (idx >= total) return;
  int n = (int)(idx >> 8);
  dst[idx] = (cnt[n] > 0) ? 0.f : src[idx];
}

__global__ __launch_bounds__(256)
void scatter_rows_kernel(float* __restrict__ dst, const float* __restrict__ src,
                         const int* __restrict__ edges, const int* __restrict__ oiList,
                         const float* __restrict__ sm) {
  int ei = blockIdx.x;
  int e  = oiList ? oiList[ei] : ei;
  float w = sm[e];
  int s = edges[e * 8 + 6], d = edges[e * 8 + 7];
  atomicAdd(&dst[(size_t)d * DIM + threadIdx.x],
            w * src[(size_t)s * DIM + threadIdx.x]);
}

// ---------- final in-place linear: emb = lrelu(emb @ Wlin + blin) ----------

__global__ __launch_bounds__(256)
void linear_kernel(float* __restrict__ embio, const bf16_t* __restrict__ WT,
                   const float* __restrict__ bias) {
  __shared__ __align__(16) bf16_t sA[16 * 256];
  const int tid = threadIdx.x, lane = tid & 31, wave = tid >> 5;
  const size_t rBase = (size_t)blockIdx.x * 16;
  {
    int row = tid >> 4;
    int c0  = (tid & 15) * 16;
    const float* src = embio + (rBase + row) * DIM + c0;
    bf16_t* dst = sA + row * 256 + c0;
    #pragma unroll
    for (int i = 0; i < 16; i += 4) {
      float4 v = *(const float4*)(src + i);
      dst[i+0]=to_bf(v.x); dst[i+1]=to_bf(v.y); dst[i+2]=to_bf(v.z); dst[i+3]=to_bf(v.w);
    }
  }
  __syncthreads();   // rows fully staged -> in-place overwrite is safe
  #pragma unroll
  for (int nt = 0; nt < 2; ++nt) {
    const int n0 = (wave * 2 + nt) * 16;
    v8f acc = {};
    #pragma unroll
    for (int k0 = 0; k0 < 256; k0 += 32) {
      v16bf a = load_a_frag(sA, 256, k0, lane);
      v16bf b = load_b_frag(WT, 256, n0, k0, lane);
      acc = wmma_bf16(a, b, acc);
    }
    int n = n0 + (lane & 15);
    float bv = bias[n];
    int mb = (lane >> 4) * 8;
    #pragma unroll
    for (int r = 0; r < 8; ++r) {
      float x = acc[r] + bv;
      x = x > 0.f ? x : 0.01f * x;
      embio[(rBase + mb + r) * DIM + n] = x;
    }
  }
}

// ---------- orchestration ----------

extern "C" void kernel_launch(void* const* d_in, const int* in_sizes, int n_in,
                              void* d_out, int out_size, void* d_ws, size_t ws_size,
                              hipStream_t stream) {
  (void)in_sizes; (void)n_in; (void)out_size; (void)ws_size;
  const float* node_score = (const float*)d_in[1];
  const int*   edges0 = (const int*)d_in[2];
  const int*   edges1 = (const int*)d_in[3];
  const float* rel0 = (const float*)d_in[4];
  const float* rel1 = (const float*)d_in[5];
  const float* mem  = (const float*)d_in[6];
  const float* qs   = (const float*)d_in[7];
  const float* qr   = (const float*)d_in[8];
  const float* Wl = (const float*)d_in[9];   const float* bl = (const float*)d_in[10];
  const float* Wr = (const float*)d_in[11];  const float* br = (const float*)d_in[12];
  const float* Wc = (const float*)d_in[13];  const float* bc = (const float*)d_in[14];
  const float* Wlin = (const float*)d_in[15]; const float* blin = (const float*)d_in[16];

  char* w = (char*)d_ws;
  auto take = [&](size_t bytes) { char* p = w; w += (bytes + 255) & ~(size_t)255; return p; };
  bf16_t* WlT   = (bf16_t*)take((size_t)512 * 1024 * 2);
  bf16_t* WrT   = (bf16_t*)take((size_t)512 * 1024 * 2);
  bf16_t* WcT   = (bf16_t*)take((size_t)512 * 512 * 2);
  bf16_t* WlinT = (bf16_t*)take((size_t)256 * 256 * 2);
  float*    logits = (float*)take((size_t)NE * 4);
  float*    expb   = (float*)take((size_t)NE * 4);
  unsigned* segm   = (unsigned*)take((size_t)N_NODES * 4);
  float*    segsum = (float*)take((size_t)N_NODES * 4);
  float*    sm1    = (float*)take((size_t)NE * 4);
  float*    sm0    = (float*)take((size_t)NE * 4);
  float*    ta     = (float*)take((size_t)NE * 4);
  int*      oiList = (int*)take((size_t)BQ * KTOP * 4);
  int*      cnt    = (int*)take((size_t)N_NODES * 4);
  float*    emb1   = (float*)take((size_t)N_NODES * DIM * 4);

  float* outScore = (float*)d_out;
  float* outEmb   = outScore + N_NODES;
  float* outEdges = outEmb + (size_t)N_NODES * DIM;
  float* outOrig  = outEdges + (size_t)BQ * KTOP * 8;

  dim3 b256(256);

  // one-time weight convert+transpose to bf16 (stays hot in 192MB L2)
  cvt_transpose_kernel<<<(512 * 1024 + 255) / 256, b256, 0, stream>>>(Wl, WlT, 1024, 512);
  cvt_transpose_kernel<<<(512 * 1024 + 255) / 256, b256, 0, stream>>>(Wr, WrT, 1024, 512);
  cvt_transpose_kernel<<<(512 * 512 + 255) / 256, b256, 0, stream>>>(Wc, WcT, 512, 512);
  cvt_transpose_kernel<<<(256 * 256 + 255) / 256, b256, 0, stream>>>(Wlin, WlinT, 256, 256);

  // ===== stage 1: edges1 on memorized embedding =====
  att_score_kernel<<<NE / MT, b256, SMEM_TOTAL, stream>>>(edges1, rel1, mem, qs, qr,
                                                          WlT, bl, WrT, br, WcT, bc, logits);
  hipMemsetAsync(segm, 0, (size_t)N_NODES * 4, stream);
  hipMemsetAsync(segsum, 0, (size_t)N_NODES * 4, stream);
  seg_max_kernel<<<NE / 256, b256, 0, stream>>>(logits, edges1, segm, NE);
  seg_expsum_kernel<<<NE / 256, b256, 0, stream>>>(logits, edges1, segm, expb, segsum, NE);
  seg_norm_kernel<<<NE / 256, b256, 0, stream>>>(expb, edges1, segsum, sm1, NE);
  ta_kernel<<<NE / 256, b256, 0, stream>>>(sm1, edges1, node_score, ta, NE);

  hipMemsetAsync(outScore, 0, (size_t)N_NODES * 4, stream);
  hipMemsetAsync(cnt, 0, (size_t)N_NODES * 4, stream);
  topk_kernel<<<BQ, b256, 0, stream>>>(ta, sm1, edges1, outScore, outEdges, outOrig,
                                       oiList, cnt);
  emb_init_kernel<<<((size_t)N_NODES * DIM) / 256, b256, 0, stream>>>(emb1, mem, cnt,
                                                                      (size_t)N_NODES * DIM);
  scatter_rows_kernel<<<BQ * KTOP, b256, 0, stream>>>(emb1, mem, edges1, oiList, sm1);

  // ===== stage 0: edges0 on updated embedding =====
  att_score_kernel<<<NE / MT, b256, SMEM_TOTAL, stream>>>(edges0, rel0, emb1, qs, qr,
                                                          WlT, bl, WrT, br, WcT, bc, logits);
  hipMemsetAsync(segm, 0, (size_t)N_NODES * 4, stream);
  hipMemsetAsync(segsum, 0, (size_t)N_NODES * 4, stream);
  seg_max_kernel<<<NE / 256, b256, 0, stream>>>(logits, edges0, segm, NE);
  seg_expsum_kernel<<<NE / 256, b256, 0, stream>>>(logits, edges0, segm, expb, segsum, NE);
  seg_norm_kernel<<<NE / 256, b256, 0, stream>>>(expb, edges0, segsum, sm0, NE);
  hipMemsetAsync(cnt, 0, (size_t)N_NODES * 4, stream);
  count_kernel<<<NE / 256, b256, 0, stream>>>(edges0, cnt, NE);
  emb_init_kernel<<<((size_t)N_NODES * DIM) / 256, b256, 0, stream>>>(outEmb, emb1, cnt,
                                                                      (size_t)N_NODES * DIM);
  scatter_rows_kernel<<<NE, b256, 0, stream>>>(outEmb, emb1, edges0, nullptr, sm0);

  // final in-place linear + leaky_relu
  linear_kernel<<<N_NODES / 16, b256, 0, stream>>>(outEmb, WlinT, blin);
}
